// MSDeformAttn_STATrans_one_shot_51582557225485
// MI455X (gfx1250) — compile-verified, matrix operand order checked by reading, not run on hardware
//
#include <hip/hip_runtime.h>
#include <math.h>

// ---------------------------------------------------------------- constants
#define D_MODEL   256
#define N_HEADS   8
#define N_LEVELS  4
#define N_POINTS  4
#define LEN_IN    19560
#define N_BATCH   2
#define M_ROWS    (N_BATCH * LEN_IN)   // 39120

typedef __attribute__((ext_vector_type(16))) _Float16 v16h;
typedef __attribute__((ext_vector_type(8)))  _Float16 v8h;
typedef __attribute__((ext_vector_type(4)))  _Float16 v4h;
typedef __attribute__((ext_vector_type(8)))  float    v8f;
typedef __attribute__((ext_vector_type(4)))  float    v4f;

// ============================================================== GEMM (K=256)
// C[M x Ncols] = A[M x 256] @ B[256 x Ncols] (+ A2 @ B2) + bias
// fp32 in memory, f16 WMMA compute with f32 accumulate.
// Block: 256 threads = 8 waves. Block tile 128x64, K-step 32.
// If A2 != nullptr the k-loop runs a virtual K=512: second half uses A2/B2
// (used to fuse  out = o0@W0f + o1@W1f + b_f  into a single pass over C).
__global__ __launch_bounds__(256) void gemm_k256(
    const float* __restrict__ A,  const float* __restrict__ B,
    const float* __restrict__ A2, const float* __restrict__ B2,
    const float* __restrict__ bias, float* __restrict__ C,
    int M, int Ncols)
{
    constexpr int K = 256, BM = 128, BN = 64, BK = 32;
    constexpr int LDA = 40, LDB = 40;   // 20-dword row stride -> conflict-free b128 fragment loads

    __shared__ __align__(16) _Float16 As[BM * LDA];
    __shared__ __align__(16) _Float16 Bs[BN * LDB];

    const int tid  = threadIdx.x;
    const int lane = tid & 31;
    const int wave = tid >> 5;          // 0..7 -> 16-row M strip
    const int m0   = blockIdx.y * BM;
    const int n0   = blockIdx.x * BN;

    const int l16  = lane & 15;
    const int half = lane >> 4;         // half-wave selects K sub-blocks
    const int kb   = half * 8;

    // B staging map: thread owns column n, 8 consecutive k (packed b64 stores)
    const int bn = tid & 63;            // 0..63
    const int bkg = (tid >> 6) * 8;     // 0,8,16,24

    const int Ktot = A2 ? 2 * K : K;
    v8f acc0 = {}, acc1 = {}, acc2 = {}, acc3 = {};

    for (int k0 = 0; k0 < Ktot; k0 += BK) {
        const float* Ap = (k0 < K) ? A : A2;
        const float* Bp = (k0 < K) ? B : B2;
        const int kk = k0 & (K - 1);

        // ---- stage A tile (128x32 f32 -> f16): v4f load -> packed b64 LDS store
        #pragma unroll
        for (int j = 0; j < 4; ++j) {
            int s   = tid + j * 256;            // 0..1023 float4 slots
            int row = s >> 3;                   // 0..127
            int kq  = s & 7;                    // float4 group within 32
            v4f av = {};
            int gm = m0 + row;
            if (gm < M)
                av = *reinterpret_cast<const v4f*>(Ap + (size_t)gm * K + kk + kq * 4);
            v4h hv;
            hv[0] = (_Float16)av[0]; hv[1] = (_Float16)av[1];
            hv[2] = (_Float16)av[2]; hv[3] = (_Float16)av[3];
            *reinterpret_cast<v4h*>(&As[row * LDA + kq * 4]) = hv;
        }
        // ---- stage B tile (32x64 f32 -> f16, N-major): coalesced reads, 2 packed b64 stores
        {
            float bv[8];
            #pragma unroll
            for (int j = 0; j < 8; ++j)         // fixed j: 64 consecutive floats across threads
                bv[j] = Bp[(size_t)(kk + bkg + j) * Ncols + n0 + bn];
            v4h h0, h1;
            #pragma unroll
            for (int j = 0; j < 4; ++j) { h0[j] = (_Float16)bv[j]; h1[j] = (_Float16)bv[4 + j]; }
            *reinterpret_cast<v4h*>(&Bs[bn * LDB + bkg])     = h0;
            *reinterpret_cast<v4h*>(&Bs[bn * LDB + bkg + 4]) = h1;
        }
        __syncthreads();

        // ---- A fragment: lane = M (0..15), halves per documented 16-bit layout
        const int arow = wave * 16 + l16;
        v8h alo = *reinterpret_cast<const v8h*>(&As[arow * LDA + kb]);        // K kb..kb+7
        v8h ahi = *reinterpret_cast<const v8h*>(&As[arow * LDA + 16 + kb]);   // K 16+kb..
        v16h afrag = __builtin_shufflevector(alo, ahi,
                        0,1,2,3,4,5,6,7,8,9,10,11,12,13,14,15);

        #pragma unroll
        for (int nt = 0; nt < 4; ++nt) {
            const int bcol = nt * 16 + l16;
            v8h blo = *reinterpret_cast<const v8h*>(&Bs[bcol * LDB + kb]);
            v8h bhi = *reinterpret_cast<const v8h*>(&Bs[bcol * LDB + 16 + kb]);
            v16h bfrag = __builtin_shufflevector(blo, bhi,
                            0,1,2,3,4,5,6,7,8,9,10,11,12,13,14,15);
            v8f* pacc = (nt == 0) ? &acc0 : (nt == 1) ? &acc1 : (nt == 2) ? &acc2 : &acc3;
            *pacc = __builtin_amdgcn_wmma_f32_16x16x32_f16(
                        false, afrag, false, bfrag, (short)0, *pacc, false, false);
        }
        __syncthreads();
    }

    // ---- epilogue: D layout -> VGPR r holds M = r + 8*(lane>=16), lane&15 = N
    #pragma unroll
    for (int nt = 0; nt < 4; ++nt) {
        const v8f accv = (nt == 0) ? acc0 : (nt == 1) ? acc1 : (nt == 2) ? acc2 : acc3;
        const int n = n0 + nt * 16 + l16;
        const float bn_add = bias ? bias[n] : 0.f;
        #pragma unroll
        for (int r = 0; r < 8; ++r) {
            const int m = m0 + wave * 16 + r + half * 8;
            if (m < M)
                C[(size_t)m * Ncols + n] = accv[r] + bn_add;
        }
    }
}

// ============================================== joint softmax over 2x16 logits
__global__ __launch_bounds__(256) void joint_softmax(
    float* __restrict__ awl0, float* __restrict__ awl1, int rows)
{
    int t = blockIdx.x * blockDim.x + threadIdx.x;
    if (t >= rows * N_HEADS) return;
    const int h   = t & (N_HEADS - 1);
    const int row = t >> 3;
    float* p0 = awl0 + (size_t)row * 128 + h * 16;
    float* p1 = awl1 + (size_t)row * 128 + h * 16;
    float v[32], mx = -INFINITY;
    #pragma unroll
    for (int i = 0; i < 16; ++i) { v[i] = p0[i]; v[16 + i] = p1[i]; }
    #pragma unroll
    for (int i = 0; i < 32; ++i) mx = fmaxf(mx, v[i]);
    float sum = 0.f;
    #pragma unroll
    for (int i = 0; i < 32; ++i) { v[i] = __expf(v[i] - mx); sum += v[i]; }
    const float inv = 1.f / sum;
    #pragma unroll
    for (int i = 0; i < 16; ++i) { p0[i] = v[i] * inv; p1[i] = v[16 + i] * inv; }
}

// ================================== deformable sampling: 1 wave per (n,q,head)
// lane = head-channel (32). value reads are 128B coalesced and L2-resident
// (both value tensors total 80 MB < 192 MB L2).
__global__ __launch_bounds__(256) void deform_sample(
    const float* __restrict__ value,   // (N, LEN, 256)  col = h*32 + d
    const float* __restrict__ offs,    // (N, Lq, 256)   col = h*32 + lvl*8 + p*2 + c
    const float* __restrict__ aw,      // (N, Lq, 128)   col = h*16 + lvl*4 + p
    const float* __restrict__ refpts,  // (N, Lq, 4, 2)
    const int*   __restrict__ shapes,  // (4,2) = (H,W)
    const int*   __restrict__ starts,  // (4,)
    float* __restrict__ out)           // (N, Lq, 256)
{
    const int row  = blockIdx.x;            // n*Lq + q
    const int lane = threadIdx.x & 31;
    const int h    = threadIdx.x >> 5;      // 0..7
    const int n    = row / LEN_IN;

    const float* vbase = value + (size_t)n * LEN_IN * 256 + h * 32 + lane;
    const float* orow  = offs  + (size_t)row * 256 + h * 32;
    const float* arow  = aw    + (size_t)row * 128 + h * 16;
    const float* rrow  = refpts + (size_t)row * 8;

    float acc = 0.f;
    #pragma unroll
    for (int lvl = 0; lvl < N_LEVELS; ++lvl) {
        const int H = shapes[2 * lvl], W = shapes[2 * lvl + 1];
        const int s = starts[lvl];
        const float fw = (float)W, fh = (float)H;
        const float rx = rrow[2 * lvl], ry = rrow[2 * lvl + 1];
        #pragma unroll
        for (int p = 0; p < N_POINTS; ++p) {
            const float x = rx * fw + orow[lvl * 8 + p * 2 + 0] - 0.5f;
            const float y = ry * fh + orow[lvl * 8 + p * 2 + 1] - 0.5f;
            const float a = arow[lvl * 4 + p];
            const float x0f = floorf(x), y0f = floorf(y);
            const int   x0 = (int)x0f,  y0 = (int)y0f;
            const float fx = x - x0f,   fy = y - y0f;
            const float w00 = (1.f - fy) * (1.f - fx), w01 = (1.f - fy) * fx;
            const float w10 = fy * (1.f - fx),         w11 = fy * fx;
            #pragma unroll
            for (int c = 0; c < 4; ++c) {
                const int   xi = x0 + (c & 1), yi = y0 + (c >> 1);
                const float wc = (c == 0) ? w00 : (c == 1) ? w01 : (c == 2) ? w10 : w11;
                if (xi >= 0 && xi < W && yi >= 0 && yi < H)
                    acc += a * wc * vbase[(size_t)(s + yi * W + xi) * 256];
            }
        }
    }
    out[(size_t)row * 256 + h * 32 + lane] = acc;
}

// ============================= fold W_o into W_agg halves (tiny, fp32 exact)
__global__ __launch_bounds__(256) void fuse_weights(
    const float* __restrict__ Wo, const float* __restrict__ Wagg,
    float* __restrict__ W0f, float* __restrict__ W1f)
{
    const int k = blockIdx.x, j = threadIdx.x;
    float s0 = 0.f, s1 = 0.f;
    for (int c = 0; c < 256; ++c) {
        const float w = Wo[k * 256 + c];
        s0 += w * Wagg[c * 256 + j];
        s1 += w * Wagg[(256 + c) * 256 + j];
    }
    W0f[k * 256 + j] = s0;
    W1f[k * 256 + j] = s1;
}

__global__ __launch_bounds__(256) void fuse_bias(
    const float* __restrict__ bo, const float* __restrict__ Wagg,
    const float* __restrict__ bagg, float* __restrict__ bf)
{
    const int j = threadIdx.x;
    float s = bagg[j];
    for (int c = 0; c < 256; ++c)
        s += bo[c] * (Wagg[c * 256 + j] + Wagg[(256 + c) * 256 + j]);
    bf[j] = s;
}

// ================================================================= launcher
extern "C" void kernel_launch(void* const* d_in, const int* in_sizes, int n_in,
                              void* d_out, int out_size, void* d_ws, size_t ws_size,
                              hipStream_t stream)
{
    (void)in_sizes; (void)n_in; (void)out_size; (void)ws_size;
    const float* q0   = (const float*)d_in[0];
    const float* q1   = (const float*)d_in[1];
    const float* ref  = (const float*)d_in[2];
    const float* f0   = (const float*)d_in[3];
    const float* f1   = (const float*)d_in[4];
    const int*   shp  = (const int*)  d_in[5];
    const int*   st   = (const int*)  d_in[6];
    const float* Wso  = (const float*)d_in[7];
    const float* bso  = (const float*)d_in[8];
    const float* Waw  = (const float*)d_in[9];
    const float* baw  = (const float*)d_in[10];
    const float* Wv   = (const float*)d_in[11];
    const float* bv   = (const float*)d_in[12];
    const float* Wo   = (const float*)d_in[13];
    const float* bo   = (const float*)d_in[14];
    const float* Wagg = (const float*)d_in[15];
    const float* bagg = (const float*)d_in[16];
    float* out = (float*)d_out;
    float* ws  = (float*)d_ws;

    const int M = M_ROWS;
    const size_t SZ256 = (size_t)M * 256, SZ128 = (size_t)M * 128;
    float* V0   = ws;
    float* V1   = V0  + SZ256;
    float* OFF0 = V1  + SZ256;
    float* OFF1 = OFF0 + SZ256;
    float* AW0  = OFF1 + SZ256;
    float* AW1  = AW0 + SZ128;
    float* O0   = AW1 + SZ128;
    float* O1   = O0  + SZ256;
    float* W0f  = O1  + SZ256;
    float* W1f  = W0f + 256 * 256;
    float* BF   = W1f + 256 * 256;

    const dim3 blk(256);
    const dim3 g256(4, (M + 127) / 128);   // Ncols = 256
    const dim3 g128(2, (M + 127) / 128);   // Ncols = 128

    // tiny weight/bias fusion (fp32 exact, negligible cost)
    fuse_weights<<<dim3(256), blk, 0, stream>>>(Wo, Wagg, W0f, W1f);
    fuse_bias   <<<dim3(1),   blk, 0, stream>>>(bo, Wagg, bagg, BF);

    // input projections (f16 WMMA, f32 accumulate)
    gemm_k256<<<g256, blk, 0, stream>>>(f0, Wv,  nullptr, nullptr, bv,  V0,   M, 256);
    gemm_k256<<<g256, blk, 0, stream>>>(f1, Wv,  nullptr, nullptr, bv,  V1,   M, 256);
    gemm_k256<<<g256, blk, 0, stream>>>(q0, Wso, nullptr, nullptr, bso, OFF0, M, 256);
    gemm_k256<<<g256, blk, 0, stream>>>(q1, Wso, nullptr, nullptr, bso, OFF1, M, 256);
    gemm_k256<<<g128, blk, 0, stream>>>(q0, Waw, nullptr, nullptr, baw, AW0,  M, 128);
    gemm_k256<<<g128, blk, 0, stream>>>(q1, Waw, nullptr, nullptr, baw, AW1,  M, 128);

    // joint softmax over both iterations' 16 logits per (n,q,head)
    joint_softmax<<<dim3((M * N_HEADS + 255) / 256), blk, 0, stream>>>(AW0, AW1, M);

    // deformable sampling
    deform_sample<<<dim3(M), blk, 0, stream>>>(V0, OFF0, AW0, ref, shp, st, O0);
    deform_sample<<<dim3(M), blk, 0, stream>>>(V1, OFF1, AW1, ref, shp, st, O1);

    // fused output projection in ONE pass: out = o0@W0f + o1@W1f + BF (virtual K=512)
    gemm_k256<<<g256, blk, 0, stream>>>(O0, W0f, O1, W1f, BF, out, M, 256);
}
